// Attention_naive_90177133347161
// MI455X (gfx1250) — matrix operand order repaired and checked by reading, not compiled, not run on hardware
//
#include <hip/hip_runtime.h>
#include <hip/hip_bf16.h>
#include <stdint.h>

// ---------------------------------------------------------------------------
// Swin-window attention, fused, WMMA f16 (f32 accumulate) for gfx1250 (MI455X)
// B=128, N=256, C=192, H=8, hd=24 (padded to 32), nW=16
// ---------------------------------------------------------------------------

#define Bc   128
#define Nc   256
#define Cc   192
#define Hc   8
#define HD   24
#define HDP  32
#define NWc  16
#define SCALE 0.20412414523193154f   // 24^-0.5

typedef _Float16 h8   __attribute__((ext_vector_type(8)));
typedef _Float16 v16h __attribute__((ext_vector_type(16)));
typedef float    v8f  __attribute__((ext_vector_type(8)));

__device__ __forceinline__ v16h cat8(h8 lo, h8 hi) {
    return __builtin_shufflevector(lo, hi, 0,1,2,3,4,5,6,7,8,9,10,11,12,13,14,15);
}
__device__ __forceinline__ v8f wmma16(v16h a, v16h b, v8f c) {
    // D = A(16x32 f16) * B(32x16 f16) + C(16x16 f32)
    return __builtin_amdgcn_wmma_f32_16x16x32_f16(false, a, false, b, (short)0, c, false, false);
}

// ------------------------------ utility kernels ----------------------------

__global__ void zero_u32_kernel(uint32_t* p, int n) {
    for (int i = blockIdx.x * blockDim.x + threadIdx.x; i < n; i += gridDim.x * blockDim.x)
        p[i] = 0u;
}

__global__ void f32_to_f16_kernel(const float* __restrict__ in, _Float16* __restrict__ out, int n) {
    for (int i = blockIdx.x * blockDim.x + threadIdx.x; i < n; i += gridDim.x * blockDim.x)
        out[i] = (_Float16)in[i];
}

// ------------------------------ QKV projection -----------------------------
// Y = x(32768x192) @ qkv_w^T(192x576); scatter into q (scaled), k, v^T (f16,
// hd padded to 32; pads pre-zeroed). grid=(2048,9), block=128 (4 waves).
__global__ void qkv_gemm_kernel(const _Float16* __restrict__ xh,   // 32768 x 192
                                const _Float16* __restrict__ wh,   // 576 x 192
                                _Float16* __restrict__ qh,         // (B*H,256,32)
                                _Float16* __restrict__ kh,         // (B*H,256,32)
                                _Float16* __restrict__ vth)        // (B*H,32,256)
{
    const int lane = threadIdx.x & 31;
    const int wv   = threadIdx.x >> 5;
    const int lr   = lane & 15;      // A row / B,C column within tile
    const int hb   = lane >> 4;      // half-wave select
    const int mb   = blockIdx.x * 16;
    const int nb   = blockIdx.y * 64 + wv * 16;

    const _Float16* arow = xh + (mb + lr) * Cc;
    const _Float16* brow = wh + (nb + lr) * Cc;

    v8f acc = {};
#pragma unroll
    for (int ks = 0; ks < 6; ++ks) {
        // A lane layout: segments k = ks*32 + hb*8 and +16 (ISA 7.12.2)
        h8 alo = *(const h8*)(arow + ks * 32 + hb * 8);
        h8 ahi = *(const h8*)(arow + ks * 32 + 16 + hb * 8);
        // B lane layout: segments k = ks*32 + hb*8 and +16 for column lr
        h8 blo = *(const h8*)(brow + ks * 32 + hb * 8);
        h8 bhi = *(const h8*)(brow + ks * 32 + 16 + hb * 8);
        acc = wmma16(cat8(alo, ahi), cat8(blo, bhi), acc);
    }

    const int c     = nb + lr;        // output column 0..575
    const int which = c / Cc;         // 0=q 1=k 2=v
    const int cc    = c % Cc;
    const int hh    = cc / HD;
    const int d     = cc % HD;
#pragma unroll
    for (int i = 0; i < 8; ++i) {
        const int mg   = mb + i + 8 * hb;       // row in 0..32767
        const int bb   = mg >> 8;
        const int seq  = mg & 255;
        const int head = bb * Hc + hh;
        const float v  = acc[i];
        if (which == 0)      qh[(head * Nc + seq) * HDP + d] = (_Float16)(v * SCALE);
        else if (which == 1) kh[(head * Nc + seq) * HDP + d] = (_Float16)v;
        else                 vth[(head * HDP + d) * Nc + seq] = (_Float16)v;
    }
}

// ------------------------------ fused attention ----------------------------
// One block (4 waves, 128 thr) per (b,h). grid = 1024.
__global__ void attn_kernel(const _Float16* __restrict__ qh,
                            const _Float16* __restrict__ kh,
                            const _Float16* __restrict__ vth,
                            const float* __restrict__ relp,   // (8,256,256)
                            const float* __restrict__ maskp,  // (16,256,256)
                            _Float16* __restrict__ aout)      // (B,N,C) f16
{
    __shared__ _Float16 sK [Nc * HDP];        // 16 KB
    __shared__ _Float16 sVT[HDP * Nc];        // 16 KB
    __shared__ _Float16 sP [4 * 16 * Nc];     // 32 KB (per-wave P strips)

    const int bh   = blockIdx.x;
    const int b    = bh >> 3;
    const int h    = bh & 7;
    const int lane = threadIdx.x & 31;
    const int wv   = threadIdx.x >> 5;
    const int lr   = lane & 15;
    const int hb   = lane >> 4;

    // stage K and V^T for this head into LDS via CDNA5 async global->LDS DMA
    // (GLOBAL_LOAD_ASYNC_TO_LDS_B128: no VGPR round trip, tracked by ASYNCcnt)
    {
        const _Float16* gk = kh  + bh * (Nc * HDP);
        const _Float16* gv = vth + bh * (HDP * Nc);
        for (int i = threadIdx.x; i < (Nc * HDP) / 8; i += 128) {
            uint64_t gak = (uint64_t)(uintptr_t)(gk + i * 8);
            uint64_t gav = (uint64_t)(uintptr_t)(gv + i * 8);
            uint32_t lak = (uint32_t)(uintptr_t)(sK  + i * 8);  // LDS offset = addr[31:0]
            uint32_t lav = (uint32_t)(uintptr_t)(sVT + i * 8);
            asm volatile("global_load_async_to_lds_b128 %0, %1, off"
                         :: "v"(lak), "v"(gak) : "memory");
            asm volatile("global_load_async_to_lds_b128 %0, %1, off"
                         :: "v"(lav), "v"(gav) : "memory");
        }
        asm volatile("s_wait_asynccnt 0x0" ::: "memory");
    }
    __syncthreads();

    const float* rp = relp  + h * (Nc * Nc);
    const float* mp = maskp + (b & (NWc - 1)) * (Nc * Nc);
    _Float16* sPw = sP + wv * 16 * Nc;

    for (int it = 0; it < 4; ++it) {
        const int strip = wv * 4 + it;
        const int Rbase = strip * 16;

        // A = Q strip row (lane layout)
        const _Float16* qrow = qh + (bh * Nc + Rbase + lr) * HDP;
        h8 qlo = *(const h8*)(qrow + hb * 8);
        h8 qhi = *(const h8*)(qrow + 16 + hb * 8);
        const v16h aq = cat8(qlo, qhi);

        // S strip = Q * K^T  (16 x 256), single K-step (hd padded to 32)
        v8f s[16];
#pragma unroll
        for (int ct = 0; ct < 16; ++ct) {
            const _Float16* kc = sK + (ct * 16 + lr) * HDP;
            h8 b0 = *(const h8*)(kc + hb * 8);
            h8 b1 = *(const h8*)(kc + 16 + hb * 8);
            v8f z = {};
            s[ct] = wmma16(aq, cat8(b0, b1), z);
        }

        // + relative_pos[h] + mask[b % nW]   (C-layout addressing)
#pragma unroll
        for (int ct = 0; ct < 16; ++ct) {
            const int col = ct * 16 + lr;
#pragma unroll
            for (int i = 0; i < 8; ++i) {
                const int row = Rbase + i + 8 * hb;
                s[ct][i] += rp[row * Nc + col] + mp[row * Nc + col];
            }
        }

        // softmax along the row (row spans 16 lanes of a half-wave x 16 tiles)
#pragma unroll
        for (int i = 0; i < 8; ++i) {
            float m = s[0][i];
#pragma unroll
            for (int ct = 1; ct < 16; ++ct) m = fmaxf(m, s[ct][i]);
            for (int off = 1; off < 16; off <<= 1) m = fmaxf(m, __shfl_xor(m, off, 32));
            float t = 0.f;
#pragma unroll
            for (int ct = 0; ct < 16; ++ct) { float e = __expf(s[ct][i] - m); s[ct][i] = e; t += e; }
            for (int off = 1; off < 16; off <<= 1) t += __shfl_xor(t, off, 32);
            const float r = 1.0f / t;
#pragma unroll
            for (int ct = 0; ct < 16; ++ct) s[ct][i] *= r;
        }

        // write P strip (f16, row-major 16x256) to this wave's LDS region
#pragma unroll
        for (int ct = 0; ct < 16; ++ct) {
            const int col = ct * 16 + lr;
#pragma unroll
            for (int i = 0; i < 8; ++i)
                sPw[(i + 8 * hb) * Nc + col] = (_Float16)s[ct][i];
        }
        asm volatile("s_wait_dscnt 0x0" ::: "memory");   // LDS store -> load, same wave

        // O strip = P(16x256) * V(256x32): 8 k-steps x 2 n-tiles
        v8f o0 = {}, o1 = {};
#pragma unroll
        for (int ksi = 0; ksi < 8; ++ksi) {
            const _Float16* prow = sPw + lr * Nc + ksi * 32;
            h8 plo = *(const h8*)(prow + hb * 8);
            h8 phi = *(const h8*)(prow + 16 + hb * 8);
            const v16h ap = cat8(plo, phi);

            const _Float16* v0p = sVT + lr * Nc + ksi * 32;        // d = lr
            const _Float16* v1p = sVT + (16 + lr) * Nc + ksi * 32; // d = 16+lr
            h8 b00 = *(const h8*)(v0p + hb * 8);  h8 b01 = *(const h8*)(v0p + 16 + hb * 8);
            h8 b10 = *(const h8*)(v1p + hb * 8);  h8 b11 = *(const h8*)(v1p + 16 + hb * 8);
            o0 = wmma16(ap, cat8(b00, b01), o0);
            o1 = wmma16(ap, cat8(b10, b11), o1);
        }

        // store strip to (B,N,C) f16 for the projection GEMM (skip d >= 24)
#pragma unroll
        for (int i = 0; i < 8; ++i) {
            const int row = Rbase + i + 8 * hb;
            _Float16* orow = aout + (b * Nc + row) * Cc + h * HD;
            orow[lr] = (_Float16)o0[i];
            if (lr < 8) orow[16 + lr] = (_Float16)o1[i];
        }
    }
}

// ------------------------------ output projection --------------------------
// out = attn_out(32768x192) @ proj_w^T(192x192) + proj_b ; f32 out.
__global__ void proj_gemm_kernel(const _Float16* __restrict__ ah,  // 32768 x 192
                                 const _Float16* __restrict__ wh,  // 192 x 192
                                 const float* __restrict__ bias,   // 192
                                 float* __restrict__ out)          // 32768 x 192
{
    const int lane = threadIdx.x & 31;
    const int wv   = threadIdx.x >> 5;
    const int lr   = lane & 15;
    const int hb   = lane >> 4;
    const int mb   = blockIdx.x * 16;
    const int nb   = blockIdx.y * 64 + wv * 16;

    const _Float16* arow = ah + (mb + lr) * Cc;
    const _Float16* brow = wh + (nb + lr) * Cc;

    v8f acc = {};
#pragma unroll
    for (int ks = 0; ks < 6; ++ks) {
        h8 alo = *(const h8*)(arow + ks * 32 + hb * 8);
        h8 ahi = *(const h8*)(arow + ks * 32 + 16 + hb * 8);
        h8 blo = *(const h8*)(brow + ks * 32 + hb * 8);
        h8 bhi = *(const h8*)(brow + ks * 32 + 16 + hb * 8);
        acc = wmma16(cat8(alo, ahi), cat8(blo, bhi), acc);
    }

    const float bv = bias[nb + lr];
#pragma unroll
    for (int i = 0; i < 8; ++i) {
        const int mg = mb + i + 8 * hb;
        out[mg * Cc + nb + lr] = acc[i] + bv;
    }
}

// ------------------------------ launcher -----------------------------------

extern "C" void kernel_launch(void* const* d_in, const int* in_sizes, int n_in,
                              void* d_out, int out_size, void* d_ws, size_t ws_size,
                              hipStream_t stream) {
    const float* x      = (const float*)d_in[0];   // (128,256,192)
    const float* relp   = (const float*)d_in[1];   // (8,256,256)
    const float* maskp  = (const float*)d_in[2];   // (16,256,256)
    const float* qkv_w  = (const float*)d_in[3];   // (576,192)
    const float* proj_w = (const float*)d_in[4];   // (192,192)
    const float* proj_b = (const float*)d_in[5];   // (192,)
    float* out = (float*)d_out;

    // workspace layout (all 16B aligned)
    char* ws = (char*)d_ws;
    const size_t SZ_XH   = (size_t)Bc * Nc * Cc * 2;        // 12.58 MB
    const size_t SZ_WQKV = (size_t)3 * Cc * Cc * 2;         // 221 KB
    const size_t SZ_WPRJ = (size_t)Cc * Cc * 2;             // 74 KB
    const size_t SZ_QKV1 = (size_t)Bc * Hc * Nc * HDP * 2;  // 16.78 MB each
    _Float16* xh   = (_Float16*)(ws);
    _Float16* wqh  = (_Float16*)(ws + SZ_XH);
    _Float16* wph  = (_Float16*)(ws + SZ_XH + SZ_WQKV);
    _Float16* qh   = (_Float16*)(ws + SZ_XH + SZ_WQKV + SZ_WPRJ);
    _Float16* kh   = (_Float16*)(ws + SZ_XH + SZ_WQKV + SZ_WPRJ + SZ_QKV1);
    _Float16* vth  = (_Float16*)(ws + SZ_XH + SZ_WQKV + SZ_WPRJ + 2 * SZ_QKV1);
    _Float16* aout = (_Float16*)(ws + SZ_XH + SZ_WQKV + SZ_WPRJ + 3 * SZ_QKV1);

    // 1) zero q/k/v^T scratch so the hd 24->32 pad is exact zero
    {
        const int nw = (int)(3 * SZ_QKV1 / 4);
        zero_u32_kernel<<<4096, 256, 0, stream>>>((uint32_t*)qh, nw);
    }
    // 2) f32 -> f16 conversions
    f32_to_f16_kernel<<<4096, 256, 0, stream>>>(x,      xh,  Bc * Nc * Cc);
    f32_to_f16_kernel<<<432,  256, 0, stream>>>(qkv_w,  wqh, 3 * Cc * Cc);
    f32_to_f16_kernel<<<144,  256, 0, stream>>>(proj_w, wph, Cc * Cc);
    // 3) QKV projection GEMM (WMMA)
    qkv_gemm_kernel<<<dim3((Bc * Nc) / 16, (3 * Cc) / 64), 128, 0, stream>>>(xh, wqh, qh, kh, vth);
    // 4) fused attention per (b,h)
    attn_kernel<<<Bc * Hc, 128, 0, stream>>>(qh, kh, vth, relp, maskp, aout);
    // 5) output projection GEMM + bias (WMMA), f32 out
    proj_gemm_kernel<<<dim3((Bc * Nc) / 16, Cc / 64), 128, 0, stream>>>(aout, wph, proj_b, out);
}
